// TopoAlignGRU_38216619000496
// MI455X (gfx1250) — matrix operand order repaired and checked by reading, not compiled, not run on hardware
//
#include <hip/hip_runtime.h>
#include <hip/hip_bf16.h>
#include <math.h>

typedef __attribute__((ext_vector_type(16))) _Float16 v16h;
typedef __attribute__((ext_vector_type(8)))  _Float16 v8h;
typedef __attribute__((ext_vector_type(8)))  float    v8f;

#define NBUS   33
#define NEDGE  64
#define BB     32
#define WW     128
#define FEATD  8
#define HD     256
#define GHD    128
#define BWROWS 4096            // B*W
#define BWNROWS 135168         // B*W*N

// A-fragment k permutation: store chunk-k order [0..7,16..23 | 8..15,24..31]
// so lanes 0-15 read halves [0..15] and lanes 16-31 read halves [16..31],
// each a contiguous 32-byte run (2 x ds_load_b128).
__device__ __forceinline__ int aperm(int kk) {
    int g = kk >> 3;                 // 0,1,2,3 -> 0,2,1,3 (swap middle groups)
    g = ((g & 1) << 1) | (g >> 1);
    return (g << 3) | (kk & 7);
}

// ---------------------------------------------------------------------------
// degree-normalized edge weights: norm[e] = d^-1/2[src] * w * d^-1/2[dst]
// ---------------------------------------------------------------------------
__global__ void norm_kernel(const int* __restrict__ ei, const float* __restrict__ ew,
                            float* __restrict__ nrm)
{
    __shared__ float deg[NBUS];
    int t = threadIdx.x;
    if (t < NBUS) deg[t] = 0.f;
    __syncthreads();
    if (t < NEDGE) atomicAdd(&deg[ei[NEDGE + t]], ew[t]);
    __syncthreads();
    if (t < NEDGE) {
        float ds = deg[ei[t]], dd = deg[ei[NEDGE + t]];
        float is = ds > 0.f ? rsqrtf(ds) : 0.f;
        float id = dd > 0.f ? rsqrtf(dd) : 0.f;
        nrm[t] = is * ew[t] * id;
    }
}

// ---------------------------------------------------------------------------
// Pack a weight W[KxN] (f32) into f16 B-fragment order, zero-padded past K:
// out[((ct*nkc + kc)*32 + lane)*16 + e] = W[kc*32 + (lane>=16?8:0) + (e>=8?16:0) + e%8][ct*16 + lane%16]
// -> every B fragment is two coalesced global_load_b128 per lane.
// ---------------------------------------------------------------------------
__global__ void pack_b_kernel(const float* __restrict__ W, _Float16* __restrict__ out,
                              int K, int N)
{
    int nkc = (K + 31) >> 5;
    int total = (N >> 4) * nkc * 512;
    int idx = blockIdx.x * blockDim.x + threadIdx.x;
    if (idx >= total) return;
    int e    = idx & 15;
    int lane = (idx >> 4) & 31;
    int t2   = idx >> 9;
    int kc   = t2 % nkc;
    int ct   = t2 / nkc;
    int k    = (kc << 5) + ((lane & 16) ? 8 : 0) + ((e & 8) ? 16 : 0) + (e & 7);
    int n    = (ct << 4) + (lane & 15);
    out[idx] = (k < K) ? (_Float16)W[(size_t)k * N + n] : (_Float16)0.f;
}

// ---------------------------------------------------------------------------
// gin = concat(feat(8), va, vm) @ pre_w(10x128) + pre_b   (K=10: VALU kernel)
// ---------------------------------------------------------------------------
__global__ void pre_kernel(const float* __restrict__ feat, const float* __restrict__ xwls,
                           const float* __restrict__ pw, const float* __restrict__ pb,
                           float* __restrict__ G0)
{
    int idx = blockIdx.x * blockDim.x + threadIdx.x;
    if (idx >= BWNROWS * GHD) return;
    int c   = idx & (GHD - 1);
    int row = idx >> 7;
    int bw  = row / NBUS;
    int n   = row - bw * NBUS;
    float va = xwls[(size_t)bw * 66 + n];
    float vm = xwls[(size_t)bw * 66 + NBUS + n];
    float acc = pb[c];
    const float* fp = feat + (size_t)bw * FEATD;
#pragma unroll
    for (int f = 0; f < FEATD; ++f) acc += fp[f] * pw[f * GHD + c];
    acc += va * pw[8 * GHD + c] + vm * pw[9 * GHD + c];
    G0[(size_t)row * GHD + c] = acc;
}

// ---------------------------------------------------------------------------
// graph propagation: Hout[:,dst] += Hin[:,src]*norm  per (b,w) graph in LDS
// ---------------------------------------------------------------------------
__global__ __launch_bounds__(128) void prop_kernel(const float* __restrict__ Hin,
                                                   float* __restrict__ Hout,
                                                   const int* __restrict__ ei,
                                                   const float* __restrict__ nrm)
{
    __shared__ float hin[NBUS * GHD];
    __shared__ float hout[NBUS * GHD];
    __shared__ int   es[NEDGE], ed[NEDGE];
    __shared__ float en[NEDGE];
    int bw = blockIdx.x;
    int t  = threadIdx.x;
    const float* src = Hin + (size_t)bw * NBUS * GHD;
    for (int i = t; i < NBUS * GHD; i += 128) { hin[i] = src[i]; hout[i] = 0.f; }
    if (t < NEDGE) { es[t] = ei[t]; ed[t] = ei[NEDGE + t]; en[t] = nrm[t]; }
    __syncthreads();
    for (int e = 0; e < NEDGE; ++e)          // thread t owns feature column t
        hout[ed[e] * GHD + t] += hin[es[e] * GHD + t] * en[e];
    __syncthreads();
    float* dst = Hout + (size_t)bw * NBUS * GHD;
    for (int i = t; i < NBUS * GHD; i += 128) dst[i] = hout[i];
}

// ---------------------------------------------------------------------------
// WMMA GEMM with pre-packed f16 weights: C[MxN] = act(acc?C:0 + A@W + bias)
// Block tile 128x64, 4 waves; each wave: 2 row-frags x 4 col-frags = 8 WMMA.
// A staged in LDS fragment order (vectorized fast path: float4 -> v8h store);
// B fragments read directly from packed global (2 x global_load_b128).
// ---------------------------------------------------------------------------
#define MT 128
#define NT 64
#define LDP 40   // padded A row length (halves): 80B rows -> conflict-spread banks
__global__ __launch_bounds__(128) void gemm_kernel(const float* __restrict__ A,
                                                   const _Float16* __restrict__ Bp,
                                                   float* __restrict__ C,
                                                   const float* __restrict__ bias,
                                                   int M_, int K_, int N_,
                                                   int accumulate, int act)
{
    __shared__ alignas(16) _Float16 AhS[MT * LDP];   // [row][perm k(32)]
    int tid  = threadIdx.x;
    int lane = tid & 31;
    int wv   = tid >> 5;
    int rowbase = blockIdx.y * MT;
    int colbase = blockIdx.x * NT;
    int halfsel = (lane < 16) ? 0 : 16;
    int nkc = (K_ + 31) >> 5;
    int ct0 = colbase >> 4;
    bool vecOK = ((K_ & 3) == 0) && (rowbase + MT <= M_);

    v8f acc[2][4];
#pragma unroll
    for (int ri = 0; ri < 2; ++ri)
#pragma unroll
        for (int j = 0; j < 4; ++j)
            acc[ri][j] = (v8f){0.f,0.f,0.f,0.f,0.f,0.f,0.f,0.f};

    const _Float16* bpl = Bp + ((size_t)lane << 4);

    for (int kc = 0; kc < nkc; ++kc) {
        int k0 = kc << 5;
        if (vecOK && (k0 + 32 <= K_)) {
            // vector staging: thread handles 4 (row, 8-k-group) units
#pragma unroll
            for (int uu = 0; uu < 4; ++uu) {
                int u = tid + uu * 128;
                int r = u >> 2, g = u & 3;
                const float4* sp = (const float4*)(A + (size_t)(rowbase + r) * K_ + k0 + (g << 3));
                float4 f0 = sp[0], f1 = sp[1];
                int gp = ((g & 1) << 1) | (g >> 1);
                v8h hv;
                hv[0] = (_Float16)f0.x; hv[1] = (_Float16)f0.y;
                hv[2] = (_Float16)f0.z; hv[3] = (_Float16)f0.w;
                hv[4] = (_Float16)f1.x; hv[5] = (_Float16)f1.y;
                hv[6] = (_Float16)f1.z; hv[7] = (_Float16)f1.w;
                *(v8h*)&AhS[r * LDP + (gp << 3)] = hv;
            }
        } else {
            // guarded scalar staging (tail chunks / K not mult of 4)
            for (int i = tid; i < MT * 32; i += 128) {
                int r = i >> 5, kk = i & 31;
                int gm = rowbase + r, gk = k0 + kk;
                float v = (gm < M_ && gk < K_) ? A[(size_t)gm * K_ + gk] : 0.f;
                AhS[r * LDP + aperm(kk)] = (_Float16)v;
            }
        }
        __syncthreads();

        v16h a0 = *(const v16h*)&AhS[(32 * wv +      (lane & 15)) * LDP + halfsel];
        v16h a1 = *(const v16h*)&AhS[(32 * wv + 16 + (lane & 15)) * LDP + halfsel];
#pragma unroll
        for (int j = 0; j < 4; ++j) {
            v16h b = *(const v16h*)(bpl + (((size_t)(ct0 + j) * nkc + kc) << 9));
            acc[0][j] = __builtin_amdgcn_wmma_f32_16x16x32_f16(false, a0, false, b,
                                                               (short)0, acc[0][j], false, false);
            acc[1][j] = __builtin_amdgcn_wmma_f32_16x16x32_f16(false, a1, false, b,
                                                               (short)0, acc[1][j], false, false);
        }
        __syncthreads();
    }

    // epilogue: element r -> (m = 32wv + 16ri + r + (lane<16?0:8), n = 16j + lane%16)
    bool fullM = (rowbase + MT <= M_);
#pragma unroll
    for (int ri = 0; ri < 2; ++ri) {
#pragma unroll
        for (int j = 0; j < 4; ++j) {
            int gn = colbase + 16 * j + (lane & 15);
            float bv = bias ? bias[gn] : 0.f;
#pragma unroll
            for (int r = 0; r < 8; ++r) {
                int gm = rowbase + 32 * wv + 16 * ri + r + ((lane < 16) ? 0 : 8);
                if (fullM || gm < M_) {
                    size_t ci = (size_t)gm * N_ + gn;
                    float v = acc[ri][j][r] + bv;
                    if (accumulate) v += C[ci];
                    if (act == 1)      v = v > 0.f ? v : 0.f;
                    else if (act == 2) v = 1.f / (1.f + __expf(-v));
                    C[ci] = v;
                }
            }
        }
    }
}

// ---------------------------------------------------------------------------
// Persistent GRU layer: one 32-wave workgroup, 128 sequential steps.
// Each wave owns a gate triple (r/z/n columns c, 256+c, 512+c): the three
// WMMA accumulators are element-aligned -> in-lane nonlinearity.
// h lives in LDS (f16, fragment-order k layout); whh packed via pack_b_kernel.
// ---------------------------------------------------------------------------
__global__ __launch_bounds__(1024) void gru_kernel(const float* __restrict__ GI,
                                                   const _Float16* __restrict__ whhP,
                                                   const float* __restrict__ bhh,
                                                   float* __restrict__ Hout)
{
    __shared__ alignas(16) _Float16 h16[BB * HD];    // [m][perm k per 32-chunk]
    int tid  = threadIdx.x;
    int lane = tid & 31;
    int wv   = tid >> 5;
    int rt   = wv & 1;          // row tile: batches [16rt,16rt+16)
    int jt   = wv >> 1;         // 0..15 column triple
    int nl   = lane & 15;
    int cH   = jt * 16 + nl;    // hidden column 0..255
    int halfsel = (lane < 16) ? 0 : 16;

    for (int i = tid; i < BB * HD; i += 1024) h16[i] = (_Float16)0.f;
    __syncthreads();

    // packed layout (K=256 -> nkc=8, N=768): base for ct = g*16 + jt
    const _Float16* wbase = whhP + ((size_t)lane << 4);

    for (int t = 0; t < WW; ++t) {
        v8f aR = (v8f){0.f,0.f,0.f,0.f,0.f,0.f,0.f,0.f};
        v8f aZ = aR, aN = aR;
#pragma unroll
        for (int kc = 0; kc < 8; ++kc) {
            int k0 = kc << 5;
            v16h a = *(const v16h*)&h16[(rt * 16 + nl) * HD + k0 + halfsel];
            v16h bR = *(const v16h*)(wbase + (((size_t)(( 0 + jt) * 8 + kc)) << 9));
            v16h bZ = *(const v16h*)(wbase + (((size_t)((16 + jt) * 8 + kc)) << 9));
            v16h bN = *(const v16h*)(wbase + (((size_t)((32 + jt) * 8 + kc)) << 9));
            aR = __builtin_amdgcn_wmma_f32_16x16x32_f16(false, a, false, bR, (short)0, aR, false, false);
            aZ = __builtin_amdgcn_wmma_f32_16x16x32_f16(false, a, false, bZ, (short)0, aZ, false, false);
            aN = __builtin_amdgcn_wmma_f32_16x16x32_f16(false, a, false, bN, (short)0, aN, false, false);
        }
        __syncthreads();   // all waves done reading h16 for this step

        float bhr = bhh[cH], bhz = bhh[256 + cH], bhn = bhh[512 + cH];
        int hoff = ((cH >> 5) << 5) + aperm(cH & 31);   // fragment-order column slot
#pragma unroll
        for (int r = 0; r < 8; ++r) {
            int m = rt * 16 + r + ((lane < 16) ? 0 : 8);     // batch row
            size_t gib = ((size_t)m * WW + t) * 768;
            float gir = GI[gib + cH];
            float giz = GI[gib + 256 + cH];
            float gin = GI[gib + 512 + cH];
            float ghr = aR[r] + bhr;
            float ghz = aZ[r] + bhz;
            float ghn = aN[r] + bhn;
            float rg = 1.f / (1.f + __expf(-(gir + ghr)));
            float zg = 1.f / (1.f + __expf(-(giz + ghz)));
            float ng = tanhf(gin + rg * ghn);
            float hold = (float)h16[m * HD + hoff];
            float hnew = (1.f - zg) * ng + zg * hold;
            h16[m * HD + hoff] = (_Float16)hnew;
            Hout[((size_t)m * WW + t) * HD + cH] = hnew;
        }
        __syncthreads();   // h16 fully updated before next step
    }
}

// ---------------------------------------------------------------------------
// u = alpha*g_t + (1-alpha)*s_t ; alpha = sigmoid(P + Tg2[bw]) ; in-place on P
// ---------------------------------------------------------------------------
__global__ void fuse_u_kernel(float* __restrict__ P, const float* __restrict__ Tg2,
                              const float* __restrict__ gt, const float* __restrict__ st)
{
    int idx = blockIdx.x * blockDim.x + threadIdx.x;
    if (idx >= BWNROWS * HD) return;
    int c   = idx & (HD - 1);
    int row = idx >> 8;
    int bw  = row / NBUS;
    float pre   = P[idx] + Tg2[(size_t)bw * HD + c];
    float alpha = 1.f / (1.f + __expf(-pre));
    float s     = st[(size_t)bw * HD + c];
    P[idx] = alpha * gt[idx] + (1.f - alpha) * s;
}

// ---------------------------------------------------------------------------
// out = x_wls + concat(u@d2_w + d2_b)  (K=256, N=2 : VALU kernel)
// ---------------------------------------------------------------------------
__global__ void final_kernel(const float* __restrict__ V, const float* __restrict__ d2w,
                             const float* __restrict__ d2b, const float* __restrict__ xwls,
                             float* __restrict__ out)
{
    int row = blockIdx.x * blockDim.x + threadIdx.x;
    if (row >= BWNROWS) return;
    const float* vp = V + (size_t)row * HD;
    float a0 = d2b[0], a1 = d2b[1];
    for (int c = 0; c < HD; ++c) {
        float v = vp[c];
        a0 += v * d2w[c * 2 + 0];
        a1 += v * d2w[c * 2 + 1];
    }
    int bw = row / NBUS;
    int n  = row - bw * NBUS;
    out[(size_t)bw * 66 + n]        = xwls[(size_t)bw * 66 + n] + a0;
    out[(size_t)bw * 66 + NBUS + n] = xwls[(size_t)bw * 66 + NBUS + n] + a1;
}

// ---------------------------------------------------------------------------
extern "C" void kernel_launch(void* const* d_in, const int* in_sizes, int n_in,
                              void* d_out, int out_size, void* d_ws, size_t ws_size,
                              hipStream_t stream)
{
    const float* r_seq  = (const float*)d_in[0];
    const float* feat   = (const float*)d_in[1];
    const float* xwls   = (const float*)d_in[2];
    const int*   ei     = (const int*)  d_in[3];
    const float* ew     = (const float*)d_in[4];
    const float* pre_w  = (const float*)d_in[5];
    const float* pre_b  = (const float*)d_in[6];
    const float* tag_w1 = (const float*)d_in[7];
    const float* tag_b1 = (const float*)d_in[8];
    const float* tag_w2 = (const float*)d_in[9];
    const float* tag_b2 = (const float*)d_in[10];
    const float* post_w = (const float*)d_in[11];
    const float* post_b = (const float*)d_in[12];
    const float* tp_w1  = (const float*)d_in[13];
    const float* tp_b1  = (const float*)d_in[14];
    const float* tp_w2  = (const float*)d_in[15];
    const float* tp_b2  = (const float*)d_in[16];
    const float* wih0   = (const float*)d_in[17];
    const float* whh0   = (const float*)d_in[18];
    const float* bih0   = (const float*)d_in[19];
    const float* bhh0   = (const float*)d_in[20];
    const float* wih1   = (const float*)d_in[21];
    const float* whh1   = (const float*)d_in[22];
    const float* bih1   = (const float*)d_in[23];
    const float* bhh1   = (const float*)d_in[24];
    const float* fg_w   = (const float*)d_in[25];
    const float* fg_b   = (const float*)d_in[26];
    const float* fs_w   = (const float*)d_in[27];
    const float* fs_b   = (const float*)d_in[28];
    const float* gate_w = (const float*)d_in[29];
    const float* gate_b = (const float*)d_in[30];
    const float* d1_w   = (const float*)d_in[31];
    const float* d1_b   = (const float*)d_in[32];
    const float* d2_w   = (const float*)d_in[33];
    const float* d2_b   = (const float*)d_in[34];
    float* out = (float*)d_out;

    char* ws = (char*)d_ws;
    const size_t SLOT = (size_t)BWNROWS * GHD * 4;   // 69,206,016 B
    float* bufA = (float*)(ws);                       // 1 slot (G0 / g2 / g_t[0] / v[0])
    float* bufB = (float*)(ws + 1 * SLOT);            // 1 slot (prop ping / g_t[1] / v[1])
    float* bufC = (float*)(ws + 2 * SLOT);            // 1 slot (prop pong)
    float* bufD = (float*)(ws + 3 * SLOT);            // 2 slots D+E (Gacc1 / Gpost / P / u)
    size_t off = 5 * SLOT;
    float* nrm = (float*)(ws + off); off += 256;
    const size_t TSZ = (size_t)BWROWS * HD * 4;
    float* T0 = (float*)(ws + off); off += TSZ;
    float* T1 = (float*)(ws + off); off += TSZ;
    float* T2 = (float*)(ws + off); off += TSZ;
    float* T3 = (float*)(ws + off); off += TSZ;
    float* TG = (float*)(ws + off); off += (size_t)BWROWS * 768 * 4;

    // packed f16 weight arena
    _Float16* pk = (_Float16*)(ws + off);
    size_t ph = 0;
    auto alloc_pk = [&](int K, int N) {
        _Float16* p = pk + ph;
        ph += (size_t)(N >> 4) * ((K + 31) >> 5) * 512;
        return p;
    };
    auto packlaunch = [&](const float* W, _Float16* p, int K, int N) {
        int total = (N >> 4) * ((K + 31) >> 5) * 512;
        pack_b_kernel<<<(total + 255) / 256, 256, 0, stream>>>(W, p, K, N);
    };

    _Float16* PT1   = alloc_pk(128, 128); alloc_pk(128,128); alloc_pk(128,128); alloc_pk(128,128);
    _Float16* PT2   = alloc_pk(128, 128); alloc_pk(128,128); alloc_pk(128,128); alloc_pk(128,128);
    _Float16* PPOST = alloc_pk(128, 256);
    _Float16* PTP1A = alloc_pk(100, 256);
    _Float16* PTP1B = alloc_pk( 66, 256);
    _Float16* PTP2  = alloc_pk(256, 256);
    _Float16* PWIH0 = alloc_pk(256, 768);
    _Float16* PWHH0 = alloc_pk(256, 768);
    _Float16* PWIH1 = alloc_pk(256, 768);
    _Float16* PWHH1 = alloc_pk(256, 768);
    _Float16* PFG   = alloc_pk(256, 256);
    _Float16* PFS   = alloc_pk(256, 256);
    _Float16* PG0   = alloc_pk(256, 256);
    _Float16* PG1   = alloc_pk(256, 256);
    _Float16* PD1   = alloc_pk(256, 256);

    const size_t S128 = 16384;   // packed halves of a 128x128 slice
    for (int k = 0; k < 4; ++k) packlaunch(tag_w1 + k * 16384, PT1 + k * S128, 128, 128);
    for (int k = 0; k < 4; ++k) packlaunch(tag_w2 + k * 16384, PT2 + k * S128, 128, 128);
    packlaunch(post_w, PPOST, 128, 256);
    packlaunch(tp_w1,             PTP1A, 100, 256);
    packlaunch(tp_w1 + 100 * 256, PTP1B,  66, 256);
    packlaunch(tp_w2, PTP2, 256, 256);
    packlaunch(wih0, PWIH0, 256, 768);
    packlaunch(whh0, PWHH0, 256, 768);
    packlaunch(wih1, PWIH1, 256, 768);
    packlaunch(whh1, PWHH1, 256, 768);
    packlaunch(fg_w, PFG, 256, 256);
    packlaunch(fs_w, PFS, 256, 256);
    packlaunch(gate_w,             PG0, 256, 256);
    packlaunch(gate_w + 256 * 256, PG1, 256, 256);
    packlaunch(d1_w, PD1, 256, 256);

    dim3 blk(128);
    dim3 gBig128(GHD / NT, BWNROWS / MT);
    dim3 gBig256(HD / NT,  BWNROWS / MT);
    dim3 gSm256(HD / NT,   BWROWS / MT);
    dim3 gSm768(768 / NT,  BWROWS / MT);

    norm_kernel<<<1, 64, 0, stream>>>(ei, ew, nrm);
    pre_kernel<<<(BWNROWS * GHD + 255) / 256, 256, 0, stream>>>(feat, xwls, pre_w, pre_b, bufA);

    // ---- TAG layer 1: Gacc(bufD) = sum_k prop^k(G0) @ W1[k] ; +bias, ReLU
    gemm_kernel<<<gBig128, blk, 0, stream>>>(bufA, PT1 + 0 * S128, bufD, nullptr, BWNROWS, 128, 128, 0, 0);
    prop_kernel<<<BWROWS, 128, 0, stream>>>(bufA, bufB, ei, nrm);
    gemm_kernel<<<gBig128, blk, 0, stream>>>(bufB, PT1 + 1 * S128, bufD, nullptr, BWNROWS, 128, 128, 1, 0);
    prop_kernel<<<BWROWS, 128, 0, stream>>>(bufB, bufC, ei, nrm);
    gemm_kernel<<<gBig128, blk, 0, stream>>>(bufC, PT1 + 2 * S128, bufD, nullptr, BWNROWS, 128, 128, 1, 0);
    prop_kernel<<<BWROWS, 128, 0, stream>>>(bufC, bufB, ei, nrm);
    gemm_kernel<<<gBig128, blk, 0, stream>>>(bufB, PT1 + 3 * S128, bufD, tag_b1, BWNROWS, 128, 128, 1, 1);

    // ---- TAG layer 2: g2(bufA) = sum_k prop^k(Gacc) @ W2[k] ; +bias
    gemm_kernel<<<gBig128, blk, 0, stream>>>(bufD, PT2 + 0 * S128, bufA, nullptr, BWNROWS, 128, 128, 0, 0);
    prop_kernel<<<BWROWS, 128, 0, stream>>>(bufD, bufB, ei, nrm);
    gemm_kernel<<<gBig128, blk, 0, stream>>>(bufB, PT2 + 1 * S128, bufA, nullptr, BWNROWS, 128, 128, 1, 0);
    prop_kernel<<<BWROWS, 128, 0, stream>>>(bufB, bufC, ei, nrm);
    gemm_kernel<<<gBig128, blk, 0, stream>>>(bufC, PT2 + 2 * S128, bufA, nullptr, BWNROWS, 128, 128, 1, 0);
    prop_kernel<<<BWROWS, 128, 0, stream>>>(bufC, bufB, ei, nrm);
    gemm_kernel<<<gBig128, blk, 0, stream>>>(bufB, PT2 + 3 * S128, bufA, tag_b2, BWNROWS, 128, 128, 1, 0);

    // ---- post: Gpost(bufD, 2 slots) = g2 @ post_w + post_b
    gemm_kernel<<<gBig256, blk, 0, stream>>>(bufA, PPOST, bufD, post_b, BWNROWS, 128, 256, 0, 0);

    // ---- temporal path: split-K concat GEMM for tp1, then tp2
    gemm_kernel<<<gSm256, blk, 0, stream>>>(r_seq, PTP1A, T0, nullptr, BWROWS, 100, 256, 0, 0);
    gemm_kernel<<<gSm256, blk, 0, stream>>>(xwls,  PTP1B, T0, tp_b1,   BWROWS,  66, 256, 1, 1);
    gemm_kernel<<<gSm256, blk, 0, stream>>>(T0, PTP2, T1, tp_b2, BWROWS, 256, 256, 0, 1);

    // ---- GRU layer 0: GI precomputed as one big GEMM, then persistent scan
    gemm_kernel<<<gSm768, blk, 0, stream>>>(T1, PWIH0, TG, bih0, BWROWS, 256, 768, 0, 0);
    gru_kernel<<<1, 1024, 0, stream>>>(TG, PWHH0, bhh0, T0);
    // ---- GRU layer 1
    gemm_kernel<<<gSm768, blk, 0, stream>>>(T0, PWIH1, TG, bih1, BWROWS, 256, 768, 0, 0);
    gru_kernel<<<1, 1024, 0, stream>>>(TG, PWHH1, bhh1, T1);

    // ---- s_t and its gate contribution (rows 256..511 of gate_w)
    gemm_kernel<<<gSm256, blk, 0, stream>>>(T1, PFS, T2, fs_b,   BWROWS, 256, 256, 0, 0);
    gemm_kernel<<<gSm256, blk, 0, stream>>>(T2, PG1, T3, gate_b, BWROWS, 256, 256, 0, 0);

    // ---- fusion head
    gemm_kernel<<<gBig256, blk, 0, stream>>>(bufD, PFG, bufA, fg_b,    BWNROWS, 256, 256, 0, 0); // g_t -> [A,B]
    gemm_kernel<<<gBig256, blk, 0, stream>>>(bufA, PG0, bufD, nullptr, BWNROWS, 256, 256, 0, 0); // P -> [D,E]
    fuse_u_kernel<<<(BWNROWS * HD + 255) / 256, 256, 0, stream>>>(bufD, T3, bufA, T2);           // u in-place
    gemm_kernel<<<gBig256, blk, 0, stream>>>(bufD, PD1, bufA, d1_b, BWNROWS, 256, 256, 0, 1);    // v = relu(u@d1)
    final_kernel<<<(BWNROWS + 255) / 256, 256, 0, stream>>>(bufA, d2_w, d2_b, xwls, out);
}
